// SingleHeadCausalSelfAttention_17798344475091
// MI455X (gfx1250) — matrix operand order repaired and checked by reading, not compiled, not run on hardware
//
#include <hip/hip_runtime.h>

typedef __attribute__((ext_vector_type(16))) _Float16 v16h;
typedef __attribute__((ext_vector_type(8)))  _Float16 v8h;
typedef __attribute__((ext_vector_type(8)))  float    v8f;
typedef __attribute__((ext_vector_type(4)))  unsigned v4u;
typedef __attribute__((ext_vector_type(8)))  int      v8i;
typedef __attribute__((ext_vector_type(4)))  int      v4i;

#define BATCH 4
#define SEQ   4096
#define DIM   256
#define ROWS  (BATCH*SEQ)     // 16384
#define P_STRIDE 72           // halfs; LDS row stride for P transpose tile

// LDS layout for attn (dynamic shared):
//   K tile:  64 rows x (512B data + 16B pad)  = 33792 B, x2 buffers
//   Vt tile: 256 rows x (128B data + 16B pad) = 36864 B, x2 buffers
//   P tile:  4 waves x 16 x P_STRIDE halfs    =  9216 B
#define K_ROW_B   528
#define V_ROW_B   144
#define K_BUF_B   (64 * K_ROW_B)      // 33792
#define V_BUF_B   (256 * V_ROW_B)     // 36864
#define SM_K(buf) ((buf) * K_BUF_B)
#define SM_V(buf) (2 * K_BUF_B + (buf) * V_BUF_B)
#define SM_P      (2 * K_BUF_B + 2 * V_BUF_B)
#define SM_TOTAL  (SM_P + 4 * 16 * P_STRIDE * 2)   // 150528 B

union V16U { v16h v; v8h h[2]; };

// A-fragment (16x32 f16, M x K): lanes<16 hold K={k..k+7, k+16..k+23},
// lanes>=16 hold K={k+8..k+15, k+24..k+31}. Two contiguous 16B loads.
__device__ __forceinline__ v16h ldA(const _Float16* row_k, int hi) {
  V16U u;
  u.h[0] = *(const v8h*)(row_k + hi * 8);
  u.h[1] = *(const v8h*)(row_k + hi * 8 + 16);
  return u.v;
}

// B-fragment (32x16 f16, K x N): lane = column n; lanes<16 hold K={k..k+7, k+16..k+23},
// lanes>=16 hold K={k+8..k+15, k+24..k+31}. Two 16B reads of the transposed row.
__device__ __forceinline__ v16h ldB(const _Float16* btrow_k, int hi) {
  V16U u;
  u.h[0] = *(const v8h*)(btrow_k + hi * 8);
  u.h[1] = *(const v8h*)(btrow_k + hi * 8 + 16);
  return u.v;
}

// ---- Tensor Data Mover: async 2D tile Global->LDS (6-arg clang-23 builtin) ----
// D# group0: [1:0]=count=1, [63:32]=lds_addr, [120:64]=global_addr, [127:126]=type=2
// D# group1: data_size=2B, pad_enable + pad_interval/pad_amount (LDS row padding
//            to dodge bank conflicts), tensor_dim0/1, tile_dim0/1, dim0 stride.
__device__ __forceinline__ void tdm_load_2d(unsigned lds_off, const void* gptr,
                                            unsigned td0, unsigned td1,
                                            unsigned tile0, unsigned tile1,
                                            unsigned stride0,
                                            unsigned pad_icode, unsigned pad_acode) {
  unsigned long long ga = (unsigned long long)(uintptr_t)gptr;
  v4u g0 = { 0x1u, lds_off, (unsigned)ga,
             (unsigned)((ga >> 32) & 0x01FFFFFFu) | 0x80000000u };
  v8i g1;
  g1[0] = (int)((1u << 16) | (1u << 20) | (pad_icode << 22) | (pad_acode << 25));
  g1[1] = (int)((td0 & 0xFFFFu) << 16);
  g1[2] = (int)(((td0 >> 16) & 0xFFFFu) | ((td1 & 0xFFFFu) << 16));
  g1[3] = (int)(((td1 >> 16) & 0xFFFFu) | ((tile0 & 0xFFFFu) << 16));
  g1[4] = (int)(tile1 & 0xFFFFu);
  g1[5] = (int)stride0;
  g1[6] = 0;
  g1[7] = 0;
  v4i g2 = {0, 0, 0, 0};
  v4i g3 = {0, 0, 0, 0};
  v8i g4 = {0, 0, 0, 0, 0, 0, 0, 0};
  __builtin_amdgcn_tensor_load_to_lds(g0, g1, g2, g3, g4, 0);
}

__global__ void cvt_f32_f16(const float* __restrict__ in, _Float16* __restrict__ out, int n) {
  int i = blockIdx.x * blockDim.x + threadIdx.x;
  int stride = gridDim.x * blockDim.x;
  for (; i < n; i += stride) out[i] = (_Float16)in[i];
}

// Phase 1: Q = x Wq^T (scaled 1/sqrt(D)), K = x Wk^T, V^T stored as [b][d][s].
__global__ __launch_bounds__(128) void qkv_proj(const _Float16* __restrict__ x16,
                                                const _Float16* __restrict__ w16,
                                                _Float16* __restrict__ q16,
                                                _Float16* __restrict__ k16,
                                                _Float16* __restrict__ vt16) {
  const int lane = threadIdx.x & 31;
  const int wave = threadIdx.x >> 5;
  const int lm   = lane & 15;
  const int hi   = lane >> 4;
  const int sel  = blockIdx.y;                 // 0=Q, 1=K, 2=V
  const _Float16* w = w16 + (size_t)sel * DIM * DIM;

  const int mrow = blockIdx.x * 16 + lm;
  const _Float16* arow = x16 + (size_t)mrow * DIM;

  v8f vzero = {};
  v8f acc[4];
#pragma unroll
  for (int nt = 0; nt < 4; ++nt) acc[nt] = vzero;

#pragma unroll
  for (int kk = 0; kk < 8; ++kk) {
    v16h a = ldA(arow + kk * 32, hi);
#pragma unroll
    for (int nt = 0; nt < 4; ++nt) {
      int ncol = wave * 64 + nt * 16 + lm;
      v16h bf = ldB(w + (size_t)ncol * DIM + kk * 32, hi);
      acc[nt] = __builtin_amdgcn_wmma_f32_16x16x32_f16(false, a, false, bf,
                                                       (short)0, acc[nt], false, false);
    }
  }

  const float qscale = 0.0625f;
#pragma unroll
  for (int nt = 0; nt < 4; ++nt) {
    int n = wave * 64 + nt * 16 + lm;
#pragma unroll
    for (int r = 0; r < 8; ++r) {
      int m = blockIdx.x * 16 + r + hi * 8;
      float v = acc[nt][r];
      if (sel == 0) {
        q16[(size_t)m * DIM + n] = (_Float16)(v * qscale);
      } else if (sel == 1) {
        k16[(size_t)m * DIM + n] = (_Float16)v;
      } else {
        int bb = m >> 12, sIn = m & (SEQ - 1);
        vt16[(size_t)bb * DIM * SEQ + (size_t)n * SEQ + sIn] = (_Float16)v;
      }
    }
  }
}

// Phase 2: flash attention, TDM double-buffered K/Vt staging in LDS.
// Block = 4 waves x 16 query rows = 64 q rows; key tiles of 64. Trip count
// ceil((q0+16)/64) == blockIdx.x+1 is IDENTICAL for all 4 waves -> block-uniform
// loop, so __syncthreads() + shared staging is safe.
__global__ __launch_bounds__(128) void attn(const _Float16* __restrict__ q16,
                                            const _Float16* __restrict__ k16,
                                            const _Float16* __restrict__ vt16,
                                            float* __restrict__ out) {
  extern __shared__ char smem[];
  const int lane  = threadIdx.x & 31;
  const int wave  = threadIdx.x >> 5;
  const int lm    = lane & 15;
  const int hi    = lane >> 4;
  const int b     = blockIdx.y;
  const int qtile = blockIdx.x * 4 + wave;
  const int q0    = qtile * 16;
  const unsigned smbase = (unsigned)(uintptr_t)smem;   // LDS byte offset

  const _Float16* kb = k16  + (size_t)b * SEQ * DIM;
  const _Float16* vb = vt16 + (size_t)b * DIM * SEQ;

  // Q A-fragments (K=256 -> 8 chunks) live in registers for the whole kernel.
  const _Float16* qrow = q16 + (size_t)(b * SEQ + q0 + lm) * DIM;
  v16h qf[8];
#pragma unroll
  for (int kk = 0; kk < 8; ++kk) qf[kk] = ldA(qrow + kk * 32, hi);

  v8f vzero = {};
  v8f oacc[16];
#pragma unroll
  for (int f = 0; f < 16; ++f) oacc[f] = vzero;

  float mi[8], li[8];
#pragma unroll
  for (int r = 0; r < 8; ++r) { mi[r] = -1e30f; li[r] = 0.f; }

  _Float16* pw = (_Float16*)(smem + SM_P) + wave * 16 * P_STRIDE;
  const int trips = blockIdx.x + 1;

  // Prologue: DMA tile 0 into buffer 0 (wave 0 issues; descriptors are uniform).
  if (wave == 0) {
    // K tile: 64 rows x 256 halfs, row pad 16B every 512B (icode 6 = 128 dw, acode 3 = 4 dw)
    tdm_load_2d(smbase + SM_K(0), kb, DIM, SEQ, DIM, 64, DIM, 6, 3);
    // Vt tile: 256 rows x 64 halfs, row pad 16B every 128B (icode 4 = 32 dw)
    tdm_load_2d(smbase + SM_V(0), vb, SEQ, DIM, 64, DIM, SEQ, 4, 3);
    __builtin_amdgcn_s_wait_tensorcnt(0);
  }
  __syncthreads();

  for (int it = 0; it < trips; ++it) {
    const int jb  = it * 64;
    const int cur = it & 1;
    const char* ksm = smem + SM_K(cur);
    const char* vsm = smem + SM_V(cur);

    // Kick off next tile's DMA while we compute on the current one.
    if (wave == 0 && it + 1 < trips) {
      tdm_load_2d(smbase + SM_K(cur ^ 1), kb + (size_t)(jb + 64) * DIM,
                  DIM, SEQ, DIM, 64, DIM, 6, 3);
      tdm_load_2d(smbase + SM_V(cur ^ 1), vb + (jb + 64),
                  SEQ, DIM, 64, DIM, SEQ, 4, 3);
    }

    // ---- S = Q K^T : 4 key sub-tiles x 8 K-chunks = 32 WMMA (B from LDS) ----
    v8f sacc[4];
#pragma unroll
    for (int nt = 0; nt < 4; ++nt) sacc[nt] = vzero;
#pragma unroll
    for (int kk = 0; kk < 8; ++kk) {
#pragma unroll
      for (int nt = 0; nt < 4; ++nt) {
        const _Float16* kr = (const _Float16*)(ksm + (nt * 16 + lm) * K_ROW_B + kk * 64);
        v16h bf = ldB(kr, hi);
        sacc[nt] = __builtin_amdgcn_wmma_f32_16x16x32_f16(false, qf[kk], false, bf,
                                                          (short)0, sacc[nt], false, false);
      }
    }

    // ---- causal mask + online softmax ----
    float mnew[8], scal[8], rsum[8];
#pragma unroll
    for (int r = 0; r < 8; ++r) {
      int qr = q0 + r + hi * 8;
      float mx = -1e30f;
#pragma unroll
      for (int nt = 0; nt < 4; ++nt) {
        int key = jb + nt * 16 + lm;
        float s = (key <= qr) ? sacc[nt][r] : -1e30f;
        sacc[nt][r] = s;
        mx = fmaxf(mx, s);
      }
      mx = fmaxf(mx, __shfl_xor(mx, 1, 32));
      mx = fmaxf(mx, __shfl_xor(mx, 2, 32));
      mx = fmaxf(mx, __shfl_xor(mx, 4, 32));
      mx = fmaxf(mx, __shfl_xor(mx, 8, 32));
      mnew[r] = fmaxf(mi[r], mx);
      scal[r] = __expf(mi[r] - mnew[r]);
      rsum[r] = 0.f;
    }

    // p = exp(s - m_new); row-sum; C->A transpose of P through wave-private LDS
#pragma unroll
    for (int nt = 0; nt < 4; ++nt) {
#pragma unroll
      for (int r = 0; r < 8; ++r) {
        float p = __expf(sacc[nt][r] - mnew[r]);
        rsum[r] += p;
        pw[(r + hi * 8) * P_STRIDE + nt * 16 + lm] = (_Float16)p;
      }
    }
#pragma unroll
    for (int r = 0; r < 8; ++r) {
      float t = rsum[r];
      t += __shfl_xor(t, 1, 32);
      t += __shfl_xor(t, 2, 32);
      t += __shfl_xor(t, 4, 32);
      t += __shfl_xor(t, 8, 32);
      li[r] = li[r] * scal[r] + t;
      mi[r] = mnew[r];
    }

    // rescale running output
#pragma unroll
    for (int f = 0; f < 16; ++f)
#pragma unroll
      for (int r = 0; r < 8; ++r) oacc[f][r] *= scal[r];

    // drain DS counter so cross-lane LDS P writes are visible (same wave, lockstep)
    asm volatile("s_wait_dscnt 0x0" ::: "memory");

    v16h pf[2];
#pragma unroll
    for (int kk = 0; kk < 2; ++kk) pf[kk] = ldA(pw + lm * P_STRIDE + kk * 32, hi);

    // ---- O += P V : 16 d-tiles x 2 K-chunks = 32 WMMA (B from LDS Vt) ----
#pragma unroll
    for (int f = 0; f < 16; ++f) {
#pragma unroll
      for (int kk = 0; kk < 2; ++kk) {
        const _Float16* vr = (const _Float16*)(vsm + (f * 16 + lm) * V_ROW_B + kk * 64);
        v16h bf = ldB(vr, hi);
        oacc[f] = __builtin_amdgcn_wmma_f32_16x16x32_f16(false, pf[kk], false, bf,
                                                         (short)0, oacc[f], false, false);
      }
    }

    // Hand-off: DMA for the next buffer must have landed before anyone reads it,
    // and everyone must be done with this buffer before it is overwritten next iter.
    if (wave == 0 && it + 1 < trips) __builtin_amdgcn_s_wait_tensorcnt(0);
    __syncthreads();
  }

  // epilogue: divide by softmax denominator, store fp32
#pragma unroll
  for (int r = 0; r < 8; ++r) {
    float inv = 1.0f / li[r];
    int m = q0 + r + hi * 8;
    float* orow = out + (size_t)(b * SEQ + m) * DIM;
#pragma unroll
    for (int f = 0; f < 16; ++f) orow[f * 16 + lm] = oacc[f][r] * inv;
  }
}

extern "C" void kernel_launch(void* const* d_in, const int* in_sizes, int n_in,
                              void* d_out, int out_size, void* d_ws, size_t ws_size,
                              hipStream_t stream) {
  const float* x  = (const float*)d_in[0];
  const float* Wq = (const float*)d_in[1];
  const float* Wk = (const float*)d_in[2];
  const float* Wv = (const float*)d_in[3];
  float* out = (float*)d_out;

  // ws: x16 (8MB) | w16 (384KB) | q16 (8MB) | k16 (8MB) | vt16 (8MB)
  char* ws = (char*)d_ws;
  _Float16* x16 = (_Float16*)ws;
  _Float16* w16 = (_Float16*)(ws + (size_t)ROWS * DIM * 2);
  _Float16* q16 = (_Float16*)(ws + (size_t)ROWS * DIM * 2 + 3u * DIM * DIM * 2);
  _Float16* k16 = q16 + (size_t)ROWS * DIM;
  _Float16* vt16 = k16 + (size_t)ROWS * DIM;

  cvt_f32_f16<<<2048, 256, 0, stream>>>(x, x16, ROWS * DIM);
  cvt_f32_f16<<<64, 256, 0, stream>>>(Wq, w16, DIM * DIM);
  cvt_f32_f16<<<64, 256, 0, stream>>>(Wk, w16 + DIM * DIM, DIM * DIM);
  cvt_f32_f16<<<64, 256, 0, stream>>>(Wv, w16 + 2 * DIM * DIM, DIM * DIM);

  qkv_proj<<<dim3(ROWS / 16, 3), 128, 0, stream>>>(x16, w16, q16, k16, vt16);
  attn<<<dim3(SEQ / 64, BATCH), 128, SM_TOTAL, stream>>>(q16, k16, vt16, out);
}
